// R_attention_81149112091274
// MI455X (gfx1250) — matrix operand order repaired and checked by reading, not compiled, not run on hardware
//
#include <hip/hip_runtime.h>
#include <math.h>

typedef __attribute__((ext_vector_type(8)))  float  v8f;
typedef __attribute__((ext_vector_type(16))) __bf16 v16bf;
typedef __attribute__((ext_vector_type(8)))  __bf16 v8bf;

constexpr int B  = 64;
constexpr int C  = 512;
constexpr int M  = 1024;
constexpr int KW = 5;
constexpr int PADW = 2;     // (KW-1)/2
constexpr int KC = 32;      // k-chunk staged in LDS == bf16 WMMA K depth
constexpr int LB = 40;      // bf16 row stride: 80B -> 16B-aligned b128, min bank pressure

// ---------------- Kernel 1: positional embedding table + per-channel sum ----
__global__ void pe_kernel(float* __restrict__ pe, float* __restrict__ pesum) {
    const int c = blockIdx.x;
    const int t = threadIdx.x;
    const int i = c >> 1;
    const float freq = expf(-(float)(2 * i) * (logf(10000.0f) / (float)C));
    float local = 0.0f;
#pragma unroll
    for (int j = 0; j < M / 256; ++j) {
        const int m = t + j * 256;
        const float ang = (float)m * freq;
        const float v = (c & 1) ? cosf(ang) : sinf(ang);
        pe[(size_t)c * M + m] = v;
        local += v;
    }
    __shared__ float red[256];
    red[t] = local;
    __syncthreads();
    for (int s = 128; s > 0; s >>= 1) {
        if (t < s) red[t] += red[t + s];
        __syncthreads();
    }
    if (t == 0) pesum[c] = red[0];
}

// ---------------- Kernel 2: per-(b,c) row sum of h_in over M -----------------
__global__ void rowsum_kernel(const float* __restrict__ hin, float* __restrict__ rsum) {
    const int wave = threadIdx.x >> 5;
    const int lane = threadIdx.x & 31;
    const int row  = blockIdx.x * 8 + wave;   // row in [0, B*C)
    const float4* p = (const float4*)(hin + (size_t)row * M);
    float s = 0.0f;
#pragma unroll
    for (int j = 0; j < 8; ++j) {
        const float4 v = p[j * 32 + lane];
        s += v.x + v.y + v.z + v.w;
    }
    for (int off = 16; off > 0; off >>= 1) s += __shfl_down(s, off);
    if (lane == 0) rsum[row] = s;
}

// ---------------- Kernel 3: channel conv -> gate, and rank-1 sums ------------
__global__ void gate_kernel(const float* __restrict__ rsum, const float* __restrict__ pesum,
                            const float* __restrict__ w, const float* __restrict__ bias,
                            float* __restrict__ gate, float* __restrict__ shs) {
    const int idx = blockIdx.x * 256 + threadIdx.x;   // b*C + c
    const int c = idx & (C - 1);
    const int b = idx >> 9;                           // C == 512
    const float invM = 1.0f / (float)M;
    float acc = bias[c];
#pragma unroll
    for (int k = 0; k < KW; ++k) {
        const int cc = c + k - PADW;
        if (cc >= 0 && cc < C) {
            const float y = (rsum[b * C + cc] + pesum[cc]) * invM;
            acc += y * w[c * KW + k];
        }
    }
    const float g = 1.0f / (1.0f + expf(-acc));
    gate[idx] = g;
    shs[idx]  = g * (rsum[idx] + pesum[c]);   // sum over m of hs = gate * sum(x)
}

// ---------------- Kernel 3b: one-time bf16 hi/lo split of x and h_in ---------
__global__ void split_kernel(const float* __restrict__ hin, const float* __restrict__ pe,
                             __bf16* __restrict__ xh, __bf16* __restrict__ xl,
                             __bf16* __restrict__ hh, __bf16* __restrict__ hl) {
    const int idx = blockIdx.x * 256 + threadIdx.x;       // 8 elems per thread
    const int row = idx >> 7;                             // b*C + c (M/8 = 128)
    const int c   = row & (C - 1);
    const int mo  = (idx & 127) << 3;
    const size_t gofs = (size_t)row * M + mo;
    const size_t pofs = (size_t)c * M + mo;
    v8bf vxh, vxl, vhh, vhl;
#pragma unroll
    for (int j = 0; j < 2; ++j) {
        const float4 hv = *(const float4*)&hin[gofs + j * 4];
        const float4 pv = *(const float4*)&pe[pofs + j * 4];
        const float hs4[4] = {hv.x, hv.y, hv.z, hv.w};
        const float xs4[4] = {hv.x + pv.x, hv.y + pv.y, hv.z + pv.z, hv.w + pv.w};
#pragma unroll
        for (int e = 0; e < 4; ++e) {
            const __bf16 a = (__bf16)xs4[e];
            const __bf16 b_ = (__bf16)(xs4[e] - (float)a);
            const __bf16 cc = (__bf16)hs4[e];
            const __bf16 d = (__bf16)(hs4[e] - (float)cc);
            vxh[j * 4 + e] = a; vxl[j * 4 + e] = b_;
            vhh[j * 4 + e] = cc; vhl[j * 4 + e] = d;
        }
    }
    *(v8bf*)&xh[gofs] = vxh;
    *(v8bf*)&xl[gofs] = vxl;
    *(v8bf*)&hh[gofs] = vhh;
    *(v8bf*)&hl[gofs] = vhl;
}

// ---------------- shared WMMA machinery --------------------------------------
// A fragment (16x32 bf16): lane L (r=L&15, h=L>>4) -> K = 8h+0..7 | 16+8h+0..7
__device__ __forceinline__ v16bf loadA(const __bf16 (*arr)[LB], int row, int h) {
    const v8bf p0 = *(const v8bf*)&arr[row][8 * h];
    const v8bf p1 = *(const v8bf*)&arr[row][16 + 8 * h];
    return __builtin_shufflevector(p0, p1, 0, 1, 2, 3, 4, 5, 6, 7,
                                           8, 9, 10, 11, 12, 13, 14, 15);
}
// B fragment (32x16 bf16, column N=L&15): K = 8h+0..7 | 16+8h+0..7
__device__ __forceinline__ v16bf loadB(const __bf16 (*arr)[LB], int row, int h) {
    const v8bf p0 = *(const v8bf*)&arr[row][8 * h];
    const v8bf p1 = *(const v8bf*)&arr[row][16 + 8 * h];
    return __builtin_shufflevector(p0, p1, 0, 1, 2, 3, 4, 5, 6, 7,
                                           8, 9, 10, 11, 12, 13, 14, 15);
}

__device__ __forceinline__ void splitStore(__bf16* dH, __bf16* dL, float v) {
    const __bf16 h = (__bf16)v;
    *dH = h;
    *dL = (__bf16)(v - (float)h);
}

#define WMMA_BF16(A, Bv, Cv) \
    __builtin_amdgcn_wmma_f32_16x16x32_bf16(false, (A), false, (Bv), (short)0, (Cv), false, false)

// One K=32 chunk: 8 A frags live, 4 B frags per tj, 24 WMMAs (2 Grams, split)
__device__ __forceinline__ void mma_chunk(
    const __bf16 (*XcH)[LB], const __bf16 (*XcL)[LB],
    const __bf16 (*HcH)[LB], const __bf16 (*HcL)[LB],
    const __bf16 (*XdH)[LB], const __bf16 (*XdL)[LB],
    const __bf16 (*HdH)[LB], const __bf16 (*HdL)[LB],
    int cw, int dw, int lrow, int lh, v8f gx[2][2], v8f gh[2][2]) {
    v16bf aXh[2], aXl[2], aHh[2], aHl[2];
#pragma unroll
    for (int ti = 0; ti < 2; ++ti) {
        const int r = cw + ti * 16 + lrow;
        aXh[ti] = loadA(XcH, r, lh);
        aXl[ti] = loadA(XcL, r, lh);
        aHh[ti] = loadA(HcH, r, lh);
        aHl[ti] = loadA(HcL, r, lh);
    }
#pragma unroll
    for (int tj = 0; tj < 2; ++tj) {
        const int r = dw + tj * 16 + lrow;
        const v16bf bXh = loadB(XdH, r, lh);
        const v16bf bXl = loadB(XdL, r, lh);
        const v16bf bHh = loadB(HdH, r, lh);
        const v16bf bHl = loadB(HdL, r, lh);
#pragma unroll
        for (int ti = 0; ti < 2; ++ti) {
            gx[ti][tj] = WMMA_BF16(aXh[ti], bXh, gx[ti][tj]);
            gx[ti][tj] = WMMA_BF16(aXh[ti], bXl, gx[ti][tj]);
            gx[ti][tj] = WMMA_BF16(aXl[ti], bXh, gx[ti][tj]);
            gh[ti][tj] = WMMA_BF16(aHh[ti], bHh, gh[ti][tj]);
            gh[ti][tj] = WMMA_BF16(aHh[ti], bHl, gh[ti][tj]);
            gh[ti][tj] = WMMA_BF16(aHl[ti], bHh, gh[ti][tj]);
        }
    }
}

// epilogue: cov = (gx*gc*gd + gh)/M - (shs_c*shs_d + shb_c*shb_d)/M^2 + eps*I
__device__ __forceinline__ void epilogue(
    const v8f gx[2][2], const v8f gh[2][2],
    const float* __restrict__ gate, const float* __restrict__ shs,
    const float* __restrict__ rsum, float* __restrict__ out,
    int b, int c0, int d0, int cw, int dw, int lrow, int lh, bool mirror) {
    const float invM  = 1.0f / (float)M;
    const float invM2 = invM * invM;
#pragma unroll
    for (int ti = 0; ti < 2; ++ti) {
#pragma unroll
        for (int tj = 0; tj < 2; ++tj) {
            const int dcol = d0 + dw + tj * 16 + lrow;
            const float gd = gate[b * C + dcol];
            const float sd = shs[b * C + dcol];
            const float hd = rsum[b * C + dcol];
#pragma unroll
            for (int v = 0; v < 8; ++v) {
                const int crow = c0 + cw + ti * 16 + v + (lh << 3);
                const float gc = gate[b * C + crow];
                const float sc = shs[b * C + crow];
                const float hc = rsum[b * C + crow];
                const float val = (gx[ti][tj][v] * gc * gd + gh[ti][tj][v]) * invM
                                - (sc * sd + hc * hd) * invM2
                                + ((crow == dcol) ? 1e-8f : 0.0f);
                out[((size_t)b * C + crow) * C + dcol] = val;
                if (mirror) out[((size_t)b * C + dcol) * C + crow] = val;
            }
        }
    }
}

__device__ __forceinline__ int3 tile_map(int bx) {  // upper-tri (ci,dj) of 8x8
    int t = bx, ci = 0, rl = 8;
    while (t >= rl) { t -= rl; ++ci; --rl; }
    return make_int3(ci, ci + t, 0);
}

// ---------------- Kernel 4a: gram from precomputed bf16 splits (fast path) ---
__global__ __launch_bounds__(128)
void gram_pre_kernel(const __bf16* __restrict__ xh, const __bf16* __restrict__ xl,
                     const __bf16* __restrict__ hh, const __bf16* __restrict__ hl,
                     const float* __restrict__ gate, const float* __restrict__ shs,
                     const float* __restrict__ rsum, float* __restrict__ out) {
    __shared__ __align__(16) __bf16 XcH[64][LB], XcL[64][LB];
    __shared__ __align__(16) __bf16 HcH[64][LB], HcL[64][LB];
    __shared__ __align__(16) __bf16 XdH[64][LB], XdL[64][LB];
    __shared__ __align__(16) __bf16 HdH[64][LB], HdL[64][LB];

    const int b = blockIdx.y;
    const int3 tm = tile_map(blockIdx.x);
    const int ci = tm.x, dj = tm.y;
    const int c0 = ci * 64, d0 = dj * 64;

    const int tid  = threadIdx.x;
    const int wave = tid >> 5;
    const int lane = tid & 31;
    const int cw = (wave >> 1) * 32, dw = (wave & 1) * 32;
    const int lrow = lane & 15, lh = lane >> 4;

    v8f gx[2][2] = {};
    v8f gh[2][2] = {};

    const int r0 = tid >> 2;           // 0..31
    const int g  = (tid & 3) << 3;     // 8-elem bf16 group

    for (int k0 = 0; k0 < M; k0 += KC) {
        // pure b128 copy staging: global bf16 -> LDS (no VALU conversions)
#pragma unroll
        for (int rr = 0; rr < 64; rr += 32) {
            const int r = r0 + rr;
            const size_t oc = ((size_t)b * C + c0 + r) * M + k0 + g;
            const size_t od = ((size_t)b * C + d0 + r) * M + k0 + g;
            *(v8bf*)&XcH[r][g] = *(const v8bf*)&xh[oc];
            *(v8bf*)&XcL[r][g] = *(const v8bf*)&xl[oc];
            *(v8bf*)&HcH[r][g] = *(const v8bf*)&hh[oc];
            *(v8bf*)&HcL[r][g] = *(const v8bf*)&hl[oc];
            *(v8bf*)&XdH[r][g] = *(const v8bf*)&xh[od];
            *(v8bf*)&XdL[r][g] = *(const v8bf*)&xl[od];
            *(v8bf*)&HdH[r][g] = *(const v8bf*)&hh[od];
            *(v8bf*)&HdL[r][g] = *(const v8bf*)&hl[od];
        }
        __syncthreads();
        mma_chunk(XcH, XcL, HcH, HcL, XdH, XdL, HdH, HdL, cw, dw, lrow, lh, gx, gh);
        __syncthreads();
    }
    epilogue(gx, gh, gate, shs, rsum, out, b, c0, d0, cw, dw, lrow, lh, ci != dj);
}

// ---------------- Kernel 4b: gram with in-block split (fallback, small ws) ---
__global__ __launch_bounds__(128)
void gram_kernel(const float* __restrict__ hin, const float* __restrict__ pe,
                 const float* __restrict__ gate, const float* __restrict__ shs,
                 const float* __restrict__ rsum, float* __restrict__ out) {
    __shared__ __align__(16) __bf16 XcH[64][LB], XcL[64][LB];
    __shared__ __align__(16) __bf16 HcH[64][LB], HcL[64][LB];
    __shared__ __align__(16) __bf16 XdH[64][LB], XdL[64][LB];
    __shared__ __align__(16) __bf16 HdH[64][LB], HdL[64][LB];

    const int b = blockIdx.y;
    const int3 tm = tile_map(blockIdx.x);
    const int ci = tm.x, dj = tm.y;
    const int c0 = ci * 64, d0 = dj * 64;

    const int tid  = threadIdx.x;
    const int wave = tid >> 5;
    const int lane = tid & 31;
    const int cw = (wave >> 1) * 32, dw = (wave & 1) * 32;
    const int lrow = lane & 15, lh = lane >> 4;

    v8f gx[2][2] = {};
    v8f gh[2][2] = {};

    const int r0 = tid >> 3;                  // 0..15
    const int cg = (tid & 7) << 2;            // float4 column group

    for (int k0 = 0; k0 < M; k0 += KC) {
#pragma unroll
        for (int rr = 0; rr < 64; rr += 16) {
            const int r = r0 + rr;
            {
                const float4 hv = *(const float4*)&hin[((size_t)b * C + c0 + r) * M + k0 + cg];
                const float4 pv = *(const float4*)&pe[(size_t)(c0 + r) * M + k0 + cg];
                const float xs[4] = {hv.x + pv.x, hv.y + pv.y, hv.z + pv.z, hv.w + pv.w};
                const float hs4[4] = {hv.x, hv.y, hv.z, hv.w};
#pragma unroll
                for (int e = 0; e < 4; ++e) {
                    splitStore(&XcH[r][cg + e], &XcL[r][cg + e], xs[e]);
                    splitStore(&HcH[r][cg + e], &HcL[r][cg + e], hs4[e]);
                }
            }
            {
                const float4 hv = *(const float4*)&hin[((size_t)b * C + d0 + r) * M + k0 + cg];
                const float4 pv = *(const float4*)&pe[(size_t)(d0 + r) * M + k0 + cg];
                const float xs[4] = {hv.x + pv.x, hv.y + pv.y, hv.z + pv.z, hv.w + pv.w};
                const float hs4[4] = {hv.x, hv.y, hv.z, hv.w};
#pragma unroll
                for (int e = 0; e < 4; ++e) {
                    splitStore(&XdH[r][cg + e], &XdL[r][cg + e], xs[e]);
                    splitStore(&HdH[r][cg + e], &HdL[r][cg + e], hs4[e]);
                }
            }
        }
        __syncthreads();
        mma_chunk(XcH, XcL, HcH, HcL, XdH, XdL, HdH, HdL, cw, dw, lrow, lh, gx, gh);
        __syncthreads();
    }
    epilogue(gx, gh, gate, shs, rsum, out, b, c0, d0, cw, dw, lrow, lh, ci != dj);
}

// ---------------------------------------------------------------------------
extern "C" void kernel_launch(void* const* d_in, const int* in_sizes, int n_in,
                              void* d_out, int out_size, void* d_ws, size_t ws_size,
                              hipStream_t stream) {
    const float* hin  = (const float*)d_in[0];   // (B, C, 1, M) fp32
    const float* w    = (const float*)d_in[1];   // (C, K) fp32
    const float* bias = (const float*)d_in[2];   // (C,) fp32
    float* out = (float*)d_out;                  // (B, C, C) fp32

    float* ws    = (float*)d_ws;
    float* pe    = ws;                           // C*M f32
    float* pesum = pe + (size_t)C * M;           // C
    float* rsum  = pesum + C;                    // B*C
    float* gatep = rsum + B * C;                 // B*C
    float* shsp  = gatep + B * C;                // B*C
    const size_t f32_elems = (size_t)C * M + C + 3 * (size_t)B * C;
    const size_t bf_elems  = (size_t)B * C * M;  // per bf16 array
    __bf16* xh = (__bf16*)(shsp + B * C);
    __bf16* xl = xh + bf_elems;
    __bf16* hh = xl + bf_elems;
    __bf16* hl = hh + bf_elems;
    const size_t need = f32_elems * sizeof(float) + 4 * bf_elems * sizeof(__bf16);

    pe_kernel<<<C, 256, 0, stream>>>(pe, pesum);
    rowsum_kernel<<<(B * C) / 8, 256, 0, stream>>>(hin, rsum);
    gate_kernel<<<(B * C) / 256, 256, 0, stream>>>(rsum, pesum, w, bias, gatep, shsp);

    if (ws_size >= need) {
        // fast path: one-time bf16 hi/lo split, VALU-free staging in gram
        split_kernel<<<(int)(bf_elems / 8 / 256), 256, 0, stream>>>(hin, pe, xh, xl, hh, hl);
        gram_pre_kernel<<<dim3(36, B), 128, 0, stream>>>(xh, xl, hh, hl, gatep, shsp, rsum, out);
    } else {
        gram_kernel<<<dim3(36, B), 128, 0, stream>>>(hin, pe, gatep, shsp, rsum, out);
    }
}
